// MyViT_70282844832029
// MI455X (gfx1250) — compile-verified, matrix operand order checked by reading, not run on hardware
//
#include <hip/hip_runtime.h>
#include <hip/hip_bf16.h>

// ---------------- ViT-Base forward on gfx1250 (wave32, WMMA bf16, async-LDS) ----------------
// B=32, S=197 (padded to Sp=224), D=768, L=12, H=12, DH=64, DFF=3072.
// GEMMs: v_wmma_f32_16x16x32_bf16 fed from LDS tiles staged with
// global_load_async_to_lds_b128 (ASYNCcnt) — the CDNA5 data-movement path.

#define BB   32
#define SS   197
#define SP   224          // padded tokens, multiple of 32
#define DD   768
#define LL   12
#define HH   12
#define DH   64
#define DFF  3072
#define NPAT 196          // 14*14 patches
#define EPS  1e-5f

typedef __bf16 bf16_t;
typedef __attribute__((ext_vector_type(16))) __bf16 v16bf;
typedef __attribute__((ext_vector_type(8)))  __bf16 v8bf;
typedef __attribute__((ext_vector_type(8)))  float  v8f;

// ---- CDNA5 16-bit A/B fragment loader (16x32 tile, [rows, K] with row stride ld).
// Lane L<16: row=L, K in {0..7, 16..23}; lane L>=16: row=L-16, K in {8..15, 24..31}.
__device__ __forceinline__ v16bf load_frag(const bf16_t* base, int ld) {
  const int lane = threadIdx.x & 31;
  const int row  = lane & 15;
  const int kb   = (lane >> 4) << 3;          // 0 or 8
  const bf16_t* p = base + (long)row * ld + kb;
  union { v16bf v; v8bf h[2]; } u;
  u.h[0] = *(const v8bf*)(p);
  u.h[1] = *(const v8bf*)(p + 16);
  return u.v;
}

__device__ __forceinline__ v8f wmma_bf16(v16bf a, v16bf b, v8f c) {
  return __builtin_amdgcn_wmma_f32_16x16x32_bf16(false, a, false, b, (short)0, c, false, false);
}

__device__ __forceinline__ float gelu_exact(float v) {
  return 0.5f * v * (1.0f + erff(v * 0.70710678118654752f));
}

// ---- async global -> LDS copy of a 128x64 bf16 tile (row stride ld elems).
// 256 threads: thread t covers row t>>1, 32-elem segment (t&1)*32, as 4 x b128.
__device__ __forceinline__ void async_copy_tile_128x64(const bf16_t* __restrict__ gbase,
                                                       int ld, bf16_t* lds_base) {
  const int t   = threadIdx.x;
  const int row = t >> 1;
  const int seg = (t & 1) * 32;
  const bf16_t* g = gbase + (long)row * ld + seg;
  unsigned lds_off = (unsigned)(unsigned long long)(const void*)(lds_base + row * 64 + seg);
#pragma unroll
  for (int i = 0; i < 4; ++i) {
    unsigned long long ga = (unsigned long long)(const void*)(g + i * 8);
    unsigned lo = lds_off + i * 16;
    asm volatile("global_load_async_to_lds_b128 %0, %1, off"
                 :: "v"(lo), "v"(ga) : "memory");
  }
}

__device__ __forceinline__ void wait_async_all() {
  asm volatile("s_wait_asynccnt 0x0" ::: "memory");
}

// ---------------- generic bf16 GEMM:  C = epilogue(A[M,K] @ Bm[N,K]^T + bias) ----------------
// 256 threads = 8 waves; block tile 128x128; wave tile 32(M) x 64(N); K-step 64,
// double-buffered async-LDS staging (64 KB LDS). Requires K % 64 == 0, grid covers
// M,N in 128 multiples; Mreal/Nreal guard the epilogue stores.
// mode 0: Cf = val  | mode 1: Cb = bf16(gelu(val)) | mode 2: Cf += val
// mode 3: patch-embed scatter: m = b*196+p -> x[(b*SP+1+p)*DD + n] = val
__global__ __launch_bounds__(256)
void gemm_bf16_kernel(const bf16_t* __restrict__ A, int lda,
                      const bf16_t* __restrict__ Bm, int ldb,
                      const float* __restrict__ bias,
                      float* Cf, bf16_t* Cb, long ldc,
                      int K, int Mreal, int Nreal, int mode) {
  __shared__ __align__(16) bf16_t As[2][128][64];
  __shared__ __align__(16) bf16_t Bs[2][128][64];

  const int lane = threadIdx.x & 31;
  const int wave = threadIdx.x >> 5;
  const int wm = wave >> 1, wn = wave & 1;
  const long row0b = (long)blockIdx.y * 128;
  const long col0b = (long)blockIdx.x * 128;

  const bf16_t* Ag = A  + row0b * lda;
  const bf16_t* Bg = Bm + col0b * ldb;

  // prologue: fill buffer 0
  async_copy_tile_128x64(Ag, lda, &As[0][0][0]);
  async_copy_tile_128x64(Bg, ldb, &Bs[0][0][0]);

  v8f acc[2][4] = {};
  const int nkt = K >> 6;
  for (int kt = 0; kt < nkt; ++kt) {
    const int cur = kt & 1;
    wait_async_all();
    __syncthreads();                       // buffer `cur` ready for all waves
    if (kt + 1 < nkt) {                    // prefetch next step into other buffer
      async_copy_tile_128x64(Ag + (kt + 1) * 64, lda, &As[1 - cur][0][0]);
      async_copy_tile_128x64(Bg + (kt + 1) * 64, ldb, &Bs[1 - cur][0][0]);
    }

    // preload ALL fragments of this K-step into distinct registers so the
    // scheduler can overlap LDS latency with the WMMA pipe (partial dscnt waits)
    const bf16_t* arow0 = &As[cur][wm * 32][0];
    const bf16_t* arow1 = &As[cur][wm * 32 + 16][0];
    v16bf a00 = load_frag(arow0, 64);
    v16bf a01 = load_frag(arow1, 64);
    v16bf a10 = load_frag(arow0 + 32, 64);
    v16bf a11 = load_frag(arow1 + 32, 64);
    v16bf b0[4], b1[4];
#pragma unroll
    for (int j = 0; j < 4; ++j) {
      const bf16_t* brow = &Bs[cur][wn * 64 + j * 16][0];
      b0[j] = load_frag(brow, 64);
      b1[j] = load_frag(brow + 32, 64);
    }
#pragma unroll
    for (int j = 0; j < 4; ++j) {
      acc[0][j] = wmma_bf16(a00, b0[j], acc[0][j]);
      acc[1][j] = wmma_bf16(a01, b0[j], acc[1][j]);
    }
#pragma unroll
    for (int j = 0; j < 4; ++j) {
      acc[0][j] = wmma_bf16(a10, b1[j], acc[0][j]);
      acc[1][j] = wmma_bf16(a11, b1[j], acc[1][j]);
    }
    __syncthreads();                       // all reads of `cur` done before refill
  }

  const long row0 = row0b + wm * 32;
  const long col0 = col0b + wn * 64;
  const int mo = (lane >> 4) << 3;
  const int nl = lane & 15;
#pragma unroll
  for (int i = 0; i < 2; ++i)
#pragma unroll
    for (int j = 0; j < 4; ++j)
#pragma unroll
      for (int r = 0; r < 8; ++r) {
        long m = row0 + i * 16 + r + mo;
        long n = col0 + j * 16 + nl;
        if (m >= Mreal || n >= Nreal) continue;
        float val = acc[i][j][r] + (bias ? bias[n] : 0.0f);
        if (mode == 0)      Cf[m * ldc + n] = val;
        else if (mode == 1) Cb[m * ldc + n] = (bf16_t)gelu_exact(val);
        else if (mode == 2) Cf[m * ldc + n] += val;
        else {              // patch embed scatter
          long b = m / NPAT, p = m % NPAT;
          Cf[(b * SP + 1 + p) * (long)DD + n] = val;
        }
      }
}

// ---------------- per-head QKV projection (1 wave per (b,h,32-token tile)) ----------------
// grid: (SP/32, B*H, 3) ; z selects q / k / vT. vT stored [B*H, 64, SP] (transposed).
__global__ __launch_bounds__(32)
void qkv_kernel(const bf16_t* __restrict__ hln,
                const bf16_t* __restrict__ wq, const bf16_t* __restrict__ wk,
                const bf16_t* __restrict__ wv,
                const float* __restrict__ bq, const float* __restrict__ bk,
                const float* __restrict__ bv,
                bf16_t* q, bf16_t* k, bf16_t* vT) {
  const int lane = threadIdx.x;
  const int st   = blockIdx.x * 32;
  const int bh   = blockIdx.y;
  const int b    = bh / HH, hh = bh % HH;
  const int which = blockIdx.z;

  const bf16_t* W    = (which == 0 ? wq : which == 1 ? wk : wv) + (long)hh * DH * DH;
  const float*  bias = (which == 0 ? bq : which == 1 ? bk : bv) + hh * DH;
  const bf16_t* Ab   = hln + ((long)b * SP + st) * DD + hh * DH;

  v8f acc[2][4] = {};
#pragma unroll
  for (int k0 = 0; k0 < DH; k0 += 32) {
    v16bf a0 = load_frag(Ab + k0, DD);
    v16bf a1 = load_frag(Ab + 16 * DD + k0, DD);
#pragma unroll
    for (int j = 0; j < 4; ++j) {
      v16bf bf = load_frag(W + (j * 16) * DH + k0, DH);
      acc[0][j] = wmma_bf16(a0, bf, acc[0][j]);
      acc[1][j] = wmma_bf16(a1, bf, acc[1][j]);
    }
  }

  const int mo = (lane >> 4) << 3;
  const int nl = lane & 15;
#pragma unroll
  for (int i = 0; i < 2; ++i)
#pragma unroll
    for (int j = 0; j < 4; ++j)
#pragma unroll
      for (int r = 0; r < 8; ++r) {
        int tok = st + i * 16 + r + mo;
        int e   = j * 16 + nl;
        float val = acc[i][j][r] + bias[e];
        if (which == 2)
          vT[((long)bh * DH + e) * SP + tok] = (bf16_t)val;
        else {
          bf16_t* o = (which == 0 ? q : k);
          o[((long)bh * SP + tok) * DH + e] = (bf16_t)val;
        }
      }
}

// ---------------- attention: 1 wave per (b,h,16-row tile) ----------------
// scores = qk^T*scale -> LDS -> masked softmax -> bf16 P in LDS -> o = P@V (WMMA) -> x +=
__global__ __launch_bounds__(32)
void attn_kernel(const bf16_t* __restrict__ q, const bf16_t* __restrict__ k,
                 const bf16_t* __restrict__ vT, float* __restrict__ x) {
  __shared__ __align__(16) float  sS[16 * SP];
  __shared__ __align__(16) bf16_t pS[16 * SP];
  const int lane = threadIdx.x;
  const int mt   = blockIdx.x * 16;
  const int bh   = blockIdx.y;
  const int b    = bh / HH, hh = bh % HH;

  const bf16_t* qb = q  + ((long)bh * SP + mt) * DH;
  const bf16_t* kb = k  + (long)bh * SP * DH;
  const bf16_t* vb = vT + (long)bh * DH * SP;

  v16bf qa0 = load_frag(qb, DH);
  v16bf qa1 = load_frag(qb + 32, DH);

  const int mo = (lane >> 4) << 3;
  const int nl = lane & 15;
  const float scale = 0.125f;           // 1/sqrt(64)

  for (int nt = 0; nt < SP; nt += 16) {
    v16bf kf0 = load_frag(kb + (long)nt * DH, DH);
    v16bf kf1 = load_frag(kb + (long)nt * DH + 32, DH);
    v8f s = {};
    s = wmma_bf16(qa0, kf0, s);
    s = wmma_bf16(qa1, kf1, s);
#pragma unroll
    for (int r = 0; r < 8; ++r) {
      int ml = r + mo;
      int n  = nt + nl;
      sS[ml * SP + n] = (n < SS) ? s[r] * scale : -1e30f;
    }
  }
  __syncthreads();

  if (lane < 16) {
    float mx = -1e30f;
    for (int n = 0; n < SS; ++n) mx = fmaxf(mx, sS[lane * SP + n]);
    float sum = 0.0f;
    for (int n = 0; n < SP; ++n) {
      float e = (n < SS) ? __expf(sS[lane * SP + n] - mx) : 0.0f;
      sum += e;
      sS[lane * SP + n] = e;
    }
    float inv = 1.0f / sum;
    for (int n = 0; n < SP; ++n) pS[lane * SP + n] = (bf16_t)(sS[lane * SP + n] * inv);
  }
  __syncthreads();

  v8f o[4] = {};
  for (int k0 = 0; k0 < SP; k0 += 32) {
    v16bf pa = load_frag(pS + k0, SP);           // A fragment from LDS
#pragma unroll
    for (int j = 0; j < 4; ++j) {
      v16bf vf = load_frag(vb + (long)(j * 16) * SP + k0, SP);
      o[j] = wmma_bf16(pa, vf, o[j]);
    }
  }
#pragma unroll
  for (int j = 0; j < 4; ++j)
#pragma unroll
    for (int r = 0; r < 8; ++r) {
      int tok = mt + r + mo;
      if (tok < SS)
        x[((long)b * SP + tok) * DD + hh * DH + j * 16 + nl] += o[j][r];
    }
}

// ---------------- LayerNorm (one 256-thread block per padded token row) ----------------
__global__ __launch_bounds__(256)
void ln_kernel(const float* __restrict__ x, const float* __restrict__ g,
               const float* __restrict__ bt, bf16_t* __restrict__ out) {
  const int row = blockIdx.x;
  const int s   = row % SP;
  const float* xr = x + (long)row * DD;
  bf16_t* orow = out + (long)row * DD;
  const int t = threadIdx.x;
  if (s >= SS) {                                 // zero pad rows
    for (int j = t; j < DD; j += 256) orow[j] = (bf16_t)0.0f;
    return;
  }
  __shared__ float red[256];
  float sum = 0.0f;
  for (int j = t; j < DD; j += 256) sum += xr[j];
  red[t] = sum; __syncthreads();
  for (int o = 128; o > 0; o >>= 1) { if (t < o) red[t] += red[t + o]; __syncthreads(); }
  float mean = red[0] / (float)DD; __syncthreads();
  float vs = 0.0f;
  for (int j = t; j < DD; j += 256) { float d = xr[j] - mean; vs += d * d; }
  red[t] = vs; __syncthreads();
  for (int o = 128; o > 0; o >>= 1) { if (t < o) red[t] += red[t + o]; __syncthreads(); }
  float inv = rsqrtf(red[0] / (float)DD + EPS);
  for (int j = t; j < DD; j += 256)
    orow[j] = (bf16_t)((xr[j] - mean) * inv * g[j] + bt[j]);
}

// ---------------- small elementwise kernels ----------------
__global__ void f32_to_bf16_kernel(const float* in, bf16_t* out, long n) {
  long i = (long)blockIdx.x * 256 + threadIdx.x;
  if (i < n) out[i] = (bf16_t)in[i];
}

__global__ void conv_pad_wh_kernel(const float* wh, bf16_t* out) {  // [1000,768] -> [1024,768]
  long i = (long)blockIdx.x * 256 + threadIdx.x;
  if (i >= 1024L * DD) return;
  long nrow = i / DD;
  out[i] = (nrow < 1000) ? (bf16_t)wh[i] : (bf16_t)0.0f;
}

__global__ void im2col_kernel(const float* img, bf16_t* out) {      // -> [B*196, 768] bf16
  long i = (long)blockIdx.x * 256 + threadIdx.x;
  if (i >= (long)BB * NPAT * DD) return;
  int d = (int)(i % DD);
  long t = i / DD;
  int p = (int)(t % NPAT);
  int b = (int)(t / NPAT);
  int c = d >> 8, rem = d & 255, pi = rem >> 4, pj = rem & 15;
  int row = (p / 14) * 16 + pi;
  int col = (p % 14) * 16 + pj;
  out[i] = (bf16_t)img[((long)(b * 3 + c) * 224 + row) * 224 + col];
}

__global__ void add_pos_cls_kernel(float* x, const float* cls) {
  long i = (long)blockIdx.x * 256 + threadIdx.x;
  if (i >= (long)BB * SS * DD) return;
  int d = (int)(i % DD);
  long t = i / DD;
  int s = (int)(t % SS);
  int b = (int)(t / SS);
  float ang = (float)s / powf(10000.0f, (float)(2 * (d >> 1)) / (float)DD);
  float pe  = (d & 1) ? cosf(ang) : sinf(ang);
  float* px = x + ((long)b * SP + s) * DD + d;
  *px = (s == 0) ? (cls[d] + pe) : (*px + pe);
}

__global__ void make_cls_rows_kernel(const float* x, bf16_t* out) { // [128,768], rows>=32 zero
  long i = (long)blockIdx.x * 256 + threadIdx.x;
  if (i >= 128L * DD) return;
  int r = (int)(i / DD), d = (int)(i % DD);
  out[i] = (r < BB) ? (bf16_t)x[(long)r * SP * DD + d] : (bf16_t)0.0f;
}

// ---------------- launcher ----------------
static inline long cdivl(long a, long b) { return (a + b - 1) / b; }

extern "C" void kernel_launch(void* const* d_in, const int* in_sizes, int n_in,
                              void* d_out, int out_size, void* d_ws, size_t ws_size,
                              hipStream_t stream) {
  const float* image  = (const float*)d_in[0];
  const float* W_emb  = (const float*)d_in[1];
  const float* b_emb  = (const float*)d_in[2];
  const float* cls    = (const float*)d_in[3];
  const float* Wq     = (const float*)d_in[4];  const float* bq = (const float*)d_in[5];
  const float* Wk     = (const float*)d_in[6];  const float* bk = (const float*)d_in[7];
  const float* Wv     = (const float*)d_in[8];  const float* bv = (const float*)d_in[9];
  const float* ln1g   = (const float*)d_in[10]; const float* ln1b = (const float*)d_in[11];
  const float* ln2g   = (const float*)d_in[12]; const float* ln2b = (const float*)d_in[13];
  const float* W1     = (const float*)d_in[14]; const float* b1 = (const float*)d_in[15];
  const float* W2     = (const float*)d_in[16]; const float* b2 = (const float*)d_in[17];
  const float* Wh     = (const float*)d_in[18]; const float* bh = (const float*)d_in[19];
  float* out = (float*)d_out;

  // workspace carve-up (~136 MB)
  char* ws = (char*)d_ws;
  size_t off = 0;
  auto alloc = [&](size_t bytes) -> char* {
    char* p = ws + off;
    off += (bytes + 255) & ~(size_t)255;
    return p;
  };
  float*  x       = (float*) alloc((size_t)BB * SP * DD * 4);
  bf16_t* hln     = (bf16_t*)alloc((size_t)BB * SP * DD * 2);
  bf16_t* qb      = (bf16_t*)alloc((size_t)BB * HH * SP * DH * 2);
  bf16_t* kbuf    = (bf16_t*)alloc((size_t)BB * HH * SP * DH * 2);
  bf16_t* vtb     = (bf16_t*)alloc((size_t)BB * HH * DH * SP * 2);
  bf16_t* mmid    = (bf16_t*)alloc((size_t)BB * SP * DFF * 2);
  bf16_t* patches = (bf16_t*)alloc((size_t)BB * NPAT * DD * 2);
  bf16_t* clsrows = (bf16_t*)alloc((size_t)128 * DD * 2);
  bf16_t* wEb     = (bf16_t*)alloc((size_t)DD * DD * 2);
  bf16_t* wQb     = (bf16_t*)alloc((size_t)LL * HH * DH * DH * 2);
  bf16_t* wKb     = (bf16_t*)alloc((size_t)LL * HH * DH * DH * 2);
  bf16_t* wVb     = (bf16_t*)alloc((size_t)LL * HH * DH * DH * 2);
  bf16_t* w1b     = (bf16_t*)alloc((size_t)DFF * DD * 2);
  bf16_t* w2b     = (bf16_t*)alloc((size_t)DD * DFF * 2);
  bf16_t* whb     = (bf16_t*)alloc((size_t)1024 * DD * 2);

  auto conv = [&](const float* src, bf16_t* dst, long n) {
    f32_to_bf16_kernel<<<(unsigned)cdivl(n, 256), 256, 0, stream>>>(src, dst, n);
  };

  // one-time weight conversions
  conv(W_emb, wEb, (long)DD * DD);
  conv(Wq, wQb, (long)LL * HH * DH * DH);
  conv(Wk, wKb, (long)LL * HH * DH * DH);
  conv(Wv, wVb, (long)LL * HH * DH * DH);
  conv_pad_wh_kernel<<<(unsigned)cdivl(1024L * DD, 256), 256, 0, stream>>>(Wh, whb);

  // patch embed: x[:,1:,:] = patches @ W_emb^T + b_emb  (mode 3 scatter)
  im2col_kernel<<<(unsigned)cdivl((long)BB * NPAT * DD, 256), 256, 0, stream>>>(image, patches);
  gemm_bf16_kernel<<<dim3(DD / 128, (BB * NPAT) / 128), 256, 0, stream>>>(
      patches, DD, wEb, DD, b_emb, x, nullptr, DD, DD, BB * NPAT, DD, 3);
  add_pos_cls_kernel<<<(unsigned)cdivl((long)BB * SS * DD, 256), 256, 0, stream>>>(x, cls);

  const int Mp = BB * SP;   // 7168, multiple of 128
  for (int l = 0; l < LL; ++l) {
    // --- attention block ---
    ln_kernel<<<Mp, 256, 0, stream>>>(x, ln1g + (long)l * DD, ln1b + (long)l * DD, hln);
    qkv_kernel<<<dim3(SP / 32, BB * HH, 3), 32, 0, stream>>>(
        hln,
        wQb + (long)l * HH * DH * DH, wKb + (long)l * HH * DH * DH, wVb + (long)l * HH * DH * DH,
        bq + (long)l * HH * DH, bk + (long)l * HH * DH, bv + (long)l * HH * DH,
        qb, kbuf, vtb);
    attn_kernel<<<dim3((SS + 15) / 16, BB * HH), 32, 0, stream>>>(qb, kbuf, vtb, x);

    // --- MLP block ---
    ln_kernel<<<Mp, 256, 0, stream>>>(x, ln2g + (long)l * DD, ln2b + (long)l * DD, hln);
    conv(W1 + (size_t)l * DFF * DD, w1b, (long)DFF * DD);
    gemm_bf16_kernel<<<dim3(DFF / 128, Mp / 128), 256, 0, stream>>>(
        hln, DD, w1b, DD, b1 + (long)l * DFF, nullptr, mmid, DFF, DD, Mp, DFF, 1);
    conv(W2 + (size_t)l * DD * DFF, w2b, (long)DD * DFF);
    gemm_bf16_kernel<<<dim3(DD / 128, Mp / 128), 256, 0, stream>>>(
        mmid, DFF, w2b, DFF, b2 + (long)l * DD, x, nullptr, DD, DFF, Mp, DD, 2);
  }

  // classifier head: out[32,1000] = x[:,0,:] @ Wh^T + bh
  make_cls_rows_kernel<<<(unsigned)cdivl(128L * DD, 256), 256, 0, stream>>>(x, clsrows);
  gemm_bf16_kernel<<<dim3(1024 / 128, 1), 256, 0, stream>>>(
      clsrows, DD, whb, DD, bh, out, nullptr, 1000, DD, BB, 1000, 0);
}
